// WaveguideSynth_16329465660226
// MI455X (gfx1250) — matrix operand order repaired and checked by reading, not compiled, not run on hardware
//
#include <hip/hip_runtime.h>
#include <cstdint>

#define N_SAMPLES 32768
#define N_DELAYS  512
#define N_FRAMES  16
#define BATCH     8
#define PI_F 3.14159265358979323846f

typedef _Float16 v16h __attribute__((ext_vector_type(16)));
typedef _Float16 v8h  __attribute__((ext_vector_type(8)));
typedef float    v8f  __attribute__((ext_vector_type(8)));

// ---------------------------------------------------------------------------
// Kernel 1: softmax over delays for each (frame, batch-row). Rows m>=8 are
// zero padding so the WMMA A-matrix (M=16) is branchless.
// wsm layout: [frame][m(16)][d(512)]  (f16)
// ---------------------------------------------------------------------------
__global__ __launch_bounds__(256) void wg_softmax_k(const float* __restrict__ dsel,
                                                    _Float16* __restrict__ wsm) {
    int f = blockIdx.x;     // 0..15 frame
    int m = blockIdx.y;     // 0..15 padded batch row
    int tid = threadIdx.x;  // 0..255
    _Float16* outp = wsm + (size_t)(f * 16 + m) * N_DELAYS;
    if (m >= BATCH) {
        outp[tid]        = (_Float16)0.0f;
        outp[tid + 256]  = (_Float16)0.0f;
        return;
    }
    __shared__ float red[256];
    // delay_selection: [B, 512, 16]
    float x0 = dsel[(size_t)m * (N_DELAYS * N_FRAMES) + (size_t)tid * N_FRAMES + f];
    float x1 = dsel[(size_t)m * (N_DELAYS * N_FRAMES) + (size_t)(tid + 256) * N_FRAMES + f];
    red[tid] = fmaxf(x0, x1);
    __syncthreads();
    for (int s = 128; s > 0; s >>= 1) {
        if (tid < s) red[tid] = fmaxf(red[tid], red[tid + s]);
        __syncthreads();
    }
    float mx = red[0];
    __syncthreads();
    float e0 = __expf(x0 - mx), e1 = __expf(x1 - mx);
    red[tid] = e0 + e1;
    __syncthreads();
    for (int s = 128; s > 0; s >>= 1) {
        if (tid < s) red[tid] += red[tid + s];
        __syncthreads();
    }
    float inv = __builtin_amdgcn_rcpf(red[0]);
    outp[tid]       = (_Float16)(e0 * inv);
    outp[tid + 256] = (_Float16)(e1 * inv);
}

// ---------------------------------------------------------------------------
// Kernel 2: imp_time[b][n] = lin_interp(impulse[b]^2, 32768)[n] * noise[b][n]
// ---------------------------------------------------------------------------
__global__ __launch_bounds__(256) void wg_impulse_k(const float* __restrict__ impulse,
                                                    const float* __restrict__ noise,
                                                    float* __restrict__ imp_time) {
    int idx = blockIdx.x * 256 + threadIdx.x;       // 0 .. 262143
    int b = idx >> 15;
    int n = idx & (N_SAMPLES - 1);
    float pos = ((float)n + 0.5f) * (128.0f / (float)N_SAMPLES) - 0.5f;
    pos = fminf(fmaxf(pos, 0.0f), 127.0f);
    int i0 = (int)floorf(pos);
    int i1 = min(i0 + 1, 127);
    float w = pos - (float)i0;
    float e0 = impulse[b * 128 + i0];
    float e1 = impulse[b * 128 + i1];
    float v = (e0 * e0) * (1.0f - w) + (e1 * e1) * w;
    imp_time[idx] = v * noise[idx];
}

// ---------------------------------------------------------------------------
// Kernel 3: comb GEMM with WMMA.
//   d_time[b][n] = damp_b * sum_d w[b][d][n>>11] * delays[d][n]
// Tile: M=16 (8 batches + 8 zero rows), K=512 (16 chunks of 32), N=16 samples.
// Grid: (frame=16, chunk=16), 128 threads = 4 waves, each wave does 2 N-tiles.
// ---------------------------------------------------------------------------
__global__ __launch_bounds__(128) void wg_comb_k(const _Float16* __restrict__ wsm,
                                                 const float* __restrict__ delays,
                                                 const float* __restrict__ damping,
                                                 float* __restrict__ d_time) {
    // Padded row stride 520 halves: row base = 260*m dwords -> banks 4m..4m+3,
    // so the 16 lanes' b128 loads spread over all 64 LDS banks.
    __shared__ __align__(16) _Float16 W[16 * 520];

    int f     = blockIdx.x;   // frame
    int chunk = blockIdx.y;   // 128-sample chunk inside frame
    int tid   = threadIdx.x;

    const _Float16* src = wsm + (size_t)f * 16 * N_DELAYS;
    for (int i = tid; i < 16 * N_DELAYS; i += 128) {
        int r = i >> 9, c = i & 511;
        W[r * 520 + c] = src[i];
    }
    __syncthreads();

    int wave = tid >> 5;
    int lane = tid & 31;
    int m  = lane & 15;       // A row / D column index
    int hi = lane >> 4;       // lane half
    int nbase = f * 2048 + chunk * 128 + wave * 32 + m;   // sample column, tile 0

    v8f acc0 = {};
    v8f acc1 = {};
    for (int kc = 0; kc < 16; ++kc) {
        // A fragment (16x32 f16): lane half hi holds K = 16hi + e (elems 0-15)
        // of row m.
        const _Float16* wr = &W[m * 520 + kc * 32 + hi * 16];
        v8h lo  = *(const v8h*)(wr);
        v8h hiv = *(const v8h*)(wr + 8);
        v16h a;
        #pragma unroll
        for (int e = 0; e < 8; ++e) { a[e] = lo[e]; a[e + 8] = hiv[e]; }

        // B fragment (32x16 f16): lane half hi holds K = 16hi + e, column = m.
        const float* dp = delays + (size_t)(kc * 32 + hi * 16) * N_SAMPLES + nbase;
        v16h b0, b1;
        #pragma unroll
        for (int e = 0; e < 16; ++e) {
            b0[e] = (_Float16)dp[(size_t)e * N_SAMPLES];
            b1[e] = (_Float16)dp[(size_t)e * N_SAMPLES + 16];
        }
        acc0 = __builtin_amdgcn_wmma_f32_16x16x32_f16(false, a, false, b0,
                                                      (short)0, acc0, false, false);
        acc1 = __builtin_amdgcn_wmma_f32_16x16x32_f16(false, a, false, b1,
                                                      (short)0, acc1, false, false);
    }

    // D layout: lane -> column N = m, VGPR r -> row M = r + 8*hi.
    // Rows 8..15 are zero padding; only hi==0 lanes store (batches 0..7).
    if (hi == 0) {
        #pragma unroll
        for (int r = 0; r < BATCH; ++r) {
            // sigmoid(damping)*0.9999 via fast reciprocal (1 ulp)
            float g = 0.9999f * __builtin_amdgcn_rcpf(1.0f + __expf(-damping[r]));
            d_time[(size_t)r * N_SAMPLES + nbase]      = acc0[r] * g;
            d_time[(size_t)r * N_SAMPLES + nbase + 16] = acc1[r] * g;
        }
    }
}

// ---------------------------------------------------------------------------
// Kernel 4: full spectral pipeline per batch, entirely in 256 KB LDS.
//   z = d + i*imp ; Z = FFT(z) ; untangle D,I ; S = D*I*F*N^-1.5 (Hermitian);
//   out = Re(IFFT_unnorm(S))
// Radix-2 DIF cascade with pairs of stages fused into radix-4-style passes:
// 7 fused passes + 1 twiddle-free radix-2 stage (identical ordering to 15
// radix-2 stages, ~half the LDS traffic).
// ---------------------------------------------------------------------------
__device__ __forceinline__ void wg_fft_passes(float2* Z, int tid, float sign) {
    #pragma unroll 1
    for (int p = 0; p < 7; ++p) {
        int lq = 13 - 2 * p;            // log2(quarter span): 13,11,9,7,5,3,1
        int Q  = 1 << lq;
        float ang0 = sign * PI_F / (float)(2 * Q);   // W1 = cis(ang0*j)
        for (int q = tid; q < (N_SAMPLES / 4); q += 1024) {
            int j    = q & (Q - 1);
            int base = ((q >> lq) << (lq + 2)) + j;
            float2 a = Z[base], b = Z[base + Q], c = Z[base + 2 * Q], d = Z[base + 3 * Q];
            float s1, c1;
            __sincosf(ang0 * (float)j, &s1, &c1);
            float c2 = c1 * c1 - s1 * s1;            // W2 = W1^2
            float s2 = 2.0f * c1 * s1;
            // stage A (half-span 2Q)
            float2 t0 = make_float2(a.x + c.x, a.y + c.y);
            float2 t1 = make_float2(b.x + d.x, b.y + d.y);
            float2 u0 = make_float2(a.x - c.x, a.y - c.y);
            float2 e0 = make_float2(b.x - d.x, b.y - d.y);
            float2 u  = make_float2(u0.x * c1 - u0.y * s1, u0.x * s1 + u0.y * c1);
            float2 e  = make_float2(e0.x * c1 - e0.y * s1, e0.x * s1 + e0.y * c1);
            float2 v  = make_float2(-sign * e.y, sign * e.x);   // e * (i*sign)
            // stage B (half-span Q)
            float2 r0 = make_float2(t0.x + t1.x, t0.y + t1.y);
            float2 d1 = make_float2(t0.x - t1.x, t0.y - t1.y);
            float2 r1 = make_float2(d1.x * c2 - d1.y * s2, d1.x * s2 + d1.y * c2);
            float2 r2 = make_float2(u.x + v.x, u.y + v.y);
            float2 d3 = make_float2(u.x - v.x, u.y - v.y);
            float2 r3 = make_float2(d3.x * c2 - d3.y * s2, d3.x * s2 + d3.y * c2);
            Z[base]         = r0;
            Z[base + Q]     = r1;
            Z[base + 2 * Q] = r2;
            Z[base + 3 * Q] = r3;
        }
        __syncthreads();
    }
    // final radix-2 stage (half-span 1, twiddle == 1)
    for (int t = tid; t < (N_SAMPLES / 2); t += 1024) {
        int i0 = 2 * t;
        float2 a = Z[i0], b = Z[i0 + 1];
        Z[i0]     = make_float2(a.x + b.x, a.y + b.y);
        Z[i0 + 1] = make_float2(a.x - b.x, a.y - b.y);
    }
    __syncthreads();
}

__device__ __forceinline__ void wg_bitrev(float2* Z, int tid) {
    for (int i = tid; i < N_SAMPLES; i += 1024) {
        int r = (int)(__brev((unsigned)i) >> 17);   // 15-bit reverse
        if (i < r) { float2 t = Z[i]; Z[i] = Z[r]; Z[r] = t; }
    }
    __syncthreads();
}

__device__ __forceinline__ float wg_filt_env(const float* fb, int k) {
    // sigmoid(filt) linearly upsampled 16 -> 16384 (align_corners=False)
    float pos = ((float)k + 0.5f) * (16.0f / 16384.0f) - 0.5f;
    pos = fminf(fmaxf(pos, 0.0f), 15.0f);
    int i0 = (int)floorf(pos);
    int i1 = min(i0 + 1, 15);
    float w = pos - (float)i0;
    float s0 = __builtin_amdgcn_rcpf(1.0f + __expf(-fb[i0]));
    float s1 = __builtin_amdgcn_rcpf(1.0f + __expf(-fb[i1]));
    return s0 * (1.0f - w) + s1 * w;
}

__global__ __launch_bounds__(1024) void wg_fft_k(const float* __restrict__ d_time,
                                                 const float* __restrict__ imp_time,
                                                 const float* __restrict__ filt,
                                                 float* __restrict__ out) {
    extern __shared__ float2 Z[];     // 32768 complex = 256 KB
    int b   = blockIdx.x;
    int tid = threadIdx.x;
    const float* db = d_time  + (size_t)b * N_SAMPLES;
    const float* ib = imp_time + (size_t)b * N_SAMPLES;

    for (int i = tid; i < N_SAMPLES; i += 1024)
        Z[i] = make_float2(db[i], ib[i]);
    __syncthreads();

    wg_fft_passes(Z, tid, -1.0f);     // forward DIF: natural -> bit-reversed
    wg_bitrev(Z, tid);                // -> natural-order spectrum

    // ortho norms: rfft 1/sqrt(N) twice + ortho irfft 1/sqrt(N) vs unnorm sum
    const float scale = 1.0f / (32768.0f * 181.01933598375618f);  // N^-1.5
    const float* fb = filt + b * 16;
    for (int k = tid; k <= N_SAMPLES / 2; k += 1024) {
        if (k == 0) {
            float2 z0 = Z[0];
            Z[0] = make_float2(z0.x * z0.y * wg_filt_env(fb, 0) * scale, 0.0f);
        } else if (k == N_SAMPLES / 2) {
            Z[k] = make_float2(0.0f, 0.0f);          // filt_spec pad == 0
        } else {
            float2 zk = Z[k], zm = Z[N_SAMPLES - k];
            float Dr = 0.5f * (zk.x + zm.x), Di = 0.5f * (zk.y - zm.y);
            float Ir = 0.5f * (zk.y + zm.y), Ii = 0.5f * (zm.x - zk.x);
            float F  = wg_filt_env(fb, k) * scale;
            float Sr = (Dr * Ir - Di * Ii) * F;
            float Si = (Dr * Ii + Di * Ir) * F;
            Z[k]             = make_float2(Sr,  Si);
            Z[N_SAMPLES - k] = make_float2(Sr, -Si);
        }
    }
    __syncthreads();

    wg_fft_passes(Z, tid, 1.0f);      // inverse (unnormalized)
    wg_bitrev(Z, tid);

    for (int i = tid; i < N_SAMPLES; i += 1024)
        out[(size_t)b * N_SAMPLES + i] = Z[i].x;
}

// ---------------------------------------------------------------------------
extern "C" void kernel_launch(void* const* d_in, const int* in_sizes, int n_in,
                              void* d_out, int out_size, void* d_ws, size_t ws_size,
                              hipStream_t stream) {
    const float* impulse = (const float*)d_in[0];   // [8,128]
    const float* dsel    = (const float*)d_in[1];   // [8,512,16]
    const float* damping = (const float*)d_in[2];   // [8,1]
    const float* filt    = (const float*)d_in[3];   // [8,16]
    const float* noise   = (const float*)d_in[4];   // [8,1,32768]
    const float* delays  = (const float*)d_in[5];   // [512,32768]
    float* out = (float*)d_out;                     // [8,1,32768]

    char* ws = (char*)d_ws;
    _Float16* wsm   = (_Float16*)ws;                               // 256 KB
    float* imp_time = (float*)(ws + 262144);                       // 1 MB
    float* d_time   = (float*)(ws + 262144 + 1048576);             // 1 MB

    (void)hipFuncSetAttribute(reinterpret_cast<const void*>(wg_fft_k),
                              hipFuncAttributeMaxDynamicSharedMemorySize, 262144);

    wg_softmax_k<<<dim3(16, 16), 256, 0, stream>>>(dsel, wsm);
    wg_impulse_k<<<(BATCH * N_SAMPLES) / 256, 256, 0, stream>>>(impulse, noise, imp_time);
    wg_comb_k<<<dim3(16, 16), 128, 0, stream>>>(wsm, delays, damping, d_time);
    wg_fft_k<<<BATCH, 1024, 262144, stream>>>(d_time, imp_time, filt, out);
}
